// MultiModalAttention_46076409151760
// MI455X (gfx1250) — matrix operand order repaired and checked by reading, not compiled, not run on hardware
//
#include <hip/hip_runtime.h>

// ---------------------------------------------------------------------------
// MultiModalAttention for MI455X (gfx1250, wave32, WMMA).
// All matmuls run on v_wmma_f32_16x16x32_bf16 (bf16 in, fp32 accumulate).
// Attention uses a flash-style online softmax so no [B,H,Sq,Sk] score tensor
// is ever materialized (compute-bound workload; HBM traffic stays ~0.2 GB).
// ---------------------------------------------------------------------------

typedef __bf16 bf16;
typedef __attribute__((ext_vector_type(16))) __bf16 v16bf;
typedef __attribute__((ext_vector_type(8)))  __bf16 v8bf;
typedef __attribute__((ext_vector_type(8)))  float  v8f;

#define DMODEL 1024
#define NHEAD  16
#define DK     64
#define BATCH  4
#define SQ     1024
#define SI     256
#define SEI    1024
#define QK_SCALE 0.125f        // 1/sqrt(64)
#define NEG_BIG (-1.0e30f)

// ---------------- scalar helpers ----------------
__device__ __forceinline__ bf16 f2bf(float f) {
  unsigned u = __builtin_bit_cast(unsigned, f);
  unsigned short s = (unsigned short)((u + 0x7FFFu + ((u >> 16) & 1u)) >> 16);
  return __builtin_bit_cast(bf16, s);
}
__device__ __forceinline__ bf16 to_bf(float f) { return f2bf(f); }
__device__ __forceinline__ bf16 to_bf(bf16 h)  { return h; }

__device__ __forceinline__ v16bf cat8(v8bf lo, v8bf hi) {
  return __builtin_shufflevector(lo, hi, 0,1,2,3,4,5,6,7,8,9,10,11,12,13,14,15);
}

struct F8u { union { v8f v; float f[8]; }; };

__device__ __forceinline__ float red_max16(float v) {
#pragma unroll
  for (int i = 1; i < 16; i <<= 1) v = fmaxf(v, __shfl_xor(v, i, 32));
  return v;
}
__device__ __forceinline__ float red_sum16(float v) {
#pragma unroll
  for (int i = 1; i < 16; i <<= 1) v += __shfl_xor(v, i, 32);
  return v;
}

// ---------------------------------------------------------------------------
// GEMM: Y[M,N] = X[M,K] @ W[K,N] + bias.  X fp32 or bf16, W/bias fp32.
// Block = 256 threads (8 wave32), block tile 64x128, wave tile 32x32,
// K staged through LDS in 32-chunks (bf16).  Stride 48 keeps 16B alignment.
// ---------------------------------------------------------------------------
template <typename XT, bool OUT_BF16>
__global__ __launch_bounds__(256)
void gemm_wmma(const XT* __restrict__ X, const float* __restrict__ W,
               const float* __restrict__ bias, void* __restrict__ Yv,
               int M, int N, int K) {
  __shared__ __align__(16) bf16 Xs[64 * 48];    // [row][k], k-chunk of 32
  __shared__ __align__(16) bf16 Wt[128 * 48];   // transposed: [n][k]

  const int t      = threadIdx.x;
  const int lane   = t & 31;
  const int wid    = t >> 5;
  const int half   = lane >> 4;
  const int l15    = lane & 15;
  const int wave_m = wid >> 2;            // 0..1
  const int wave_n = wid & 3;             // 0..3
  const int bm     = blockIdx.y * 64;
  const int bn     = blockIdx.x * 128;

  F8u acc[2][2];
#pragma unroll
  for (int i = 0; i < 2; ++i)
#pragma unroll
    for (int j = 0; j < 2; ++j)
      acc[i][j].v = (v8f){0.f,0.f,0.f,0.f,0.f,0.f,0.f,0.f};

  const int xr  = t >> 2;                 // 0..63
  const int xc  = (t & 3) * 8;            // 0,8,16,24
  const int wn  = t & 127;                // 0..127
  const int wkr = t >> 7;                 // 0..1

  for (int k0 = 0; k0 < K; k0 += 32) {
    __syncthreads();
    // stage X tile (fp32/bf16 -> bf16)
    const XT* xp = X + (size_t)(bm + xr) * K + k0 + xc;
#pragma unroll
    for (int i = 0; i < 8; ++i) Xs[xr * 48 + xc + i] = to_bf(xp[i]);
    // stage W tile transposed (coalesced global read along n)
#pragma unroll
    for (int kk = 0; kk < 16; ++kk) {
      int k = wkr * 16 + kk;
      Wt[wn * 48 + k] = f2bf(W[(size_t)(k0 + k) * N + bn + wn]);
    }
    __syncthreads();

    // A fragments: lane row = l15; K elems: [8h..8h+7] and [16+8h..16+8h+7]
    v16bf afr[2], bfr[2];
#pragma unroll
    for (int sm = 0; sm < 2; ++sm) {
      int row = wave_m * 32 + sm * 16 + l15;
      afr[sm] = cat8(*(const v8bf*)&Xs[row * 48 + 8 * half],
                     *(const v8bf*)&Xs[row * 48 + 16 + 8 * half]);
    }
    // B fragments: lane col = l15; K elems: [8h..8h+7] and [16+8h..16+8h+7]
#pragma unroll
    for (int sn = 0; sn < 2; ++sn) {
      int col = wave_n * 32 + sn * 16 + l15;
      bfr[sn] = cat8(*(const v8bf*)&Wt[col * 48 + 8 * half],
                     *(const v8bf*)&Wt[col * 48 + 16 + 8 * half]);
    }
#pragma unroll
    for (int sm = 0; sm < 2; ++sm)
#pragma unroll
      for (int sn = 0; sn < 2; ++sn)
        acc[sm][sn].v = __builtin_amdgcn_wmma_f32_16x16x32_bf16(
            false, afr[sm], false, bfr[sn], (short)0, acc[sm][sn].v,
            false, false);
  }

  // epilogue: C layout lane = N (l15), vgpr r -> M = r + 8*half
#pragma unroll
  for (int sm = 0; sm < 2; ++sm)
#pragma unroll
    for (int sn = 0; sn < 2; ++sn) {
      int col = bn + wave_n * 32 + sn * 16 + l15;
      float bv = bias[col];
#pragma unroll
      for (int r = 0; r < 8; ++r) {
        int row = bm + wave_m * 32 + sm * 16 + 8 * half + r;
        float v = acc[sm][sn].f[r] + bv;
        if constexpr (OUT_BF16) ((bf16*)Yv)[(size_t)row * N + col] = f2bf(v);
        else                    ((float*)Yv)[(size_t)row * N + col] = v;
      }
    }
}

// ---------------------------------------------------------------------------
// One attention phase (flash-style online softmax) for one wave's 16-query
// tile.  MODE: 0 = causal + key-padding (e), 1 = no mask (i), 2 = padding (ei).
// o/m/lsum are the phase-local running output (fp32), row-max, row-sum.
// ---------------------------------------------------------------------------
template <int MODE>
__device__ __forceinline__ void attn_phase(
    const bf16* __restrict__ Kp, const bf16* __restrict__ Vp, int SK,
    const unsigned char* __restrict__ pm, const v16bf* qa,
    int b, int h, int qbase, bf16* Ps, bf16* Vt, int lane,
    F8u* o, float* m, float* lsum) {
  const int half = lane >> 4;
  const int l15  = lane & 15;

  int kend = SK;
  if (MODE == 0) kend = ((qbase + 15) / 32 + 1) * 32;  // skip fully-causal blocks

  for (int kb = 0; kb < kend; kb += 32) {
    // prefetch next key block row (speculative, GL2)
    if (kb + 32 < kend)
      __builtin_prefetch(Kp + ((size_t)(b * SK + kb + 32 + l15) * DMODEL + h * DK), 0, 1);

    // ---- S tiles (2 n-tiles of 16 keys), Dk=64 = two K=32 WMMA steps
    F8u s[2];
#pragma unroll
    for (int nt = 0; nt < 2; ++nt) {
      int key = kb + nt * 16 + l15;
      const bf16* kr = Kp + ((size_t)(b * SK + key) * DMODEL + h * DK);
      v16bf b0 = cat8(*(const v8bf*)(kr + 8 * half),
                      *(const v8bf*)(kr + 16 + 8 * half));
      v16bf b1 = cat8(*(const v8bf*)(kr + 32 + 8 * half),
                      *(const v8bf*)(kr + 48 + 8 * half));
      v8f a = (v8f){0.f,0.f,0.f,0.f,0.f,0.f,0.f,0.f};
      a = __builtin_amdgcn_wmma_f32_16x16x32_bf16(false, qa[0], false, b0,
                                                  (short)0, a, false, false);
      a = __builtin_amdgcn_wmma_f32_16x16x32_bf16(false, qa[1], false, b1,
                                                  (short)0, a, false, false);
      s[nt].v = a;
    }

    // ---- scale + masks (C layout: lane = key%16, vgpr r -> query r+8*half)
#pragma unroll
    for (int nt = 0; nt < 2; ++nt) {
      int key  = kb + nt * 16 + l15;
      bool pad = false;
      if (MODE != 1) pad = (pm[b * SK + key] != 0);
#pragma unroll
      for (int r = 0; r < 8; ++r) {
        float v = s[nt].f[r] * QK_SCALE;
        if (MODE == 0 && key > qbase + r + 8 * half) v = NEG_BIG;
        if (pad) v = NEG_BIG;
        s[nt].f[r] = v;
      }
    }

    // ---- online softmax (row reductions across the 16-lane half-wave)
    float p0s[8], p1s[8];
#pragma unroll
    for (int r = 0; r < 8; ++r) {
      float bmax = red_max16(fmaxf(s[0].f[r], s[1].f[r]));
      float mn   = fmaxf(m[r], bmax);
      float corr = __expf(m[r] - mn);
      float p0   = __expf(s[0].f[r] - mn);
      float p1   = __expf(s[1].f[r] - mn);
      lsum[r] = lsum[r] * corr + red_sum16(p0 + p1);
      m[r] = mn;
#pragma unroll
      for (int dt = 0; dt < 4; ++dt) o[dt].f[r] *= corr;
      p0s[r] = p0; p1s[r] = p1;
    }

    // ---- restage P (C layout -> A layout) through this wave's LDS slice
#pragma unroll
    for (int r = 0; r < 8; ++r) {
      Ps[(r + 8 * half) * 48 + l15]      = f2bf(p0s[r]);
      Ps[(r + 8 * half) * 48 + 16 + l15] = f2bf(p1s[r]);
    }
    // ---- stage V block transposed: Vt[d][k]  (lane = one key row)
    {
      const bf16* vr = Vp + ((size_t)(b * SK + kb + lane) * DMODEL + h * DK);
#pragma unroll
      for (int d0 = 0; d0 < DK; d0 += 8) {
        v8bf vv = *(const v8bf*)(vr + d0);
#pragma unroll
        for (int i = 0; i < 8; ++i) Vt[(d0 + i) * 48 + lane] = vv[i];
      }
    }
    // intra-wave LDS write -> read hazard across differing address expressions
    asm volatile("s_wait_dscnt 0" ::: "memory");

    // ---- O += P @ V  (A = P[16x32], B = Vt column-slices, 4 d-tiles)
    {
      v16bf ap = cat8(*(const v8bf*)&Ps[l15 * 48 + 8 * half],
                      *(const v8bf*)&Ps[l15 * 48 + 16 + 8 * half]);
#pragma unroll
      for (int dt = 0; dt < 4; ++dt) {
        int d = dt * 16 + l15;
        v16bf bv = cat8(*(const v8bf*)&Vt[d * 48 + 8 * half],
                        *(const v8bf*)&Vt[d * 48 + 16 + 8 * half]);
        o[dt].v = __builtin_amdgcn_wmma_f32_16x16x32_bf16(
            false, ap, false, bv, (short)0, o[dt].v, false, false);
      }
    }
  }
}

// ---------------------------------------------------------------------------
// Attention kernel: one wave per (b, h, 16-query tile).  8 waves / block.
// Sums the three normalized attention outputs (lam1 = lam2 = 1) and writes
// the head-concatenated result as bf16 [B, SQ, D] for the output GEMM.
// ---------------------------------------------------------------------------
__global__ __launch_bounds__(256)
void attn_kernel(const bf16* __restrict__ Qp,
                 const bf16* __restrict__ Ke,  const bf16* __restrict__ Ve,
                 const bf16* __restrict__ Ki,  const bf16* __restrict__ Vi,
                 const bf16* __restrict__ Kei, const bf16* __restrict__ Vei,
                 const unsigned char* __restrict__ pmE,
                 const unsigned char* __restrict__ pmEI,
                 const int* __restrict__ image_bool,
                 bf16* __restrict__ Cc) {
  // per-wave slices: P [16 x 32] + Vt [64 x 32], stride 48 (16B aligned)
  __shared__ __align__(16) bf16 smem[8 * (16 * 48 + 64 * 48)];

  const int t    = threadIdx.x;
  const int lane = t & 31;
  const int wid  = t >> 5;
  const int half = lane >> 4;
  const int l15  = lane & 15;

  bf16* Ps = smem + wid * (16 * 48 + 64 * 48);
  bf16* Vt = Ps + 16 * 48;

  const int gw    = blockIdx.x * 8 + wid;   // 0..4095
  const int qt    = gw & 63;
  const int h     = (gw >> 6) & 15;
  const int b     = gw >> 10;
  const int qbase = qt * 16;

  // Q fragments kept resident for the whole pass (Dk=64 -> 2 chunks of K=32)
  v16bf qa[2];
  {
    const bf16* qr = Qp + ((size_t)(b * SQ + qbase + l15) * DMODEL + h * DK);
    qa[0] = cat8(*(const v8bf*)(qr + 8 * half),
                 *(const v8bf*)(qr + 16 + 8 * half));
    qa[1] = cat8(*(const v8bf*)(qr + 32 + 8 * half),
                 *(const v8bf*)(qr + 48 + 8 * half));
  }

  F8u ot[4], op[4];
  float m[8], ls[8];
#pragma unroll
  for (int dt = 0; dt < 4; ++dt)
    ot[dt].v = (v8f){0.f,0.f,0.f,0.f,0.f,0.f,0.f,0.f};

  const int ib = image_bool[0];

  // ---- phase e: causal + key padding
#pragma unroll
  for (int dt = 0; dt < 4; ++dt) op[dt].v = (v8f){0.f,0.f,0.f,0.f,0.f,0.f,0.f,0.f};
#pragma unroll
  for (int r = 0; r < 8; ++r) { m[r] = NEG_BIG; ls[r] = 0.f; }
  attn_phase<0>(Ke, Ve, SQ, pmE, qa, b, h, qbase, Ps, Vt, lane, op, m, ls);
#pragma unroll
  for (int dt = 0; dt < 4; ++dt)
#pragma unroll
    for (int r = 0; r < 8; ++r) ot[dt].f[r] += op[dt].f[r] / ls[r];

  if (ib != 0) {
    // ---- phase i: image keys, no mask
#pragma unroll
    for (int dt = 0; dt < 4; ++dt) op[dt].v = (v8f){0.f,0.f,0.f,0.f,0.f,0.f,0.f,0.f};
#pragma unroll
    for (int r = 0; r < 8; ++r) { m[r] = NEG_BIG; ls[r] = 0.f; }
    attn_phase<1>(Ki, Vi, SI, nullptr, qa, b, h, qbase, Ps, Vt, lane, op, m, ls);
#pragma unroll
    for (int dt = 0; dt < 4; ++dt)
#pragma unroll
      for (int r = 0; r < 8; ++r) ot[dt].f[r] += op[dt].f[r] / ls[r];

    // ---- phase ei: key padding only
#pragma unroll
    for (int dt = 0; dt < 4; ++dt) op[dt].v = (v8f){0.f,0.f,0.f,0.f,0.f,0.f,0.f,0.f};
#pragma unroll
    for (int r = 0; r < 8; ++r) { m[r] = NEG_BIG; ls[r] = 0.f; }
    attn_phase<2>(Kei, Vei, SEI, pmEI, qa, b, h, qbase, Ps, Vt, lane, op, m, ls);
#pragma unroll
    for (int dt = 0; dt < 4; ++dt)
#pragma unroll
      for (int r = 0; r < 8; ++r) ot[dt].f[r] += op[dt].f[r] / ls[r];
  }

  // ---- write concat [B, SQ, D] (head transpose is just indexing)
#pragma unroll
  for (int dt = 0; dt < 4; ++dt)
#pragma unroll
    for (int r = 0; r < 8; ++r) {
      int q = qbase + r + 8 * half;
      int d = h * DK + dt * 16 + l15;
      Cc[(size_t)(b * SQ + q) * DMODEL + d] = f2bf(ot[dt].f[r]);
    }
}

// ---------------------------------------------------------------------------
extern "C" void kernel_launch(void* const* d_in, const int* in_sizes, int n_in,
                              void* d_out, int out_size, void* d_ws, size_t ws_size,
                              hipStream_t stream) {
  (void)in_sizes; (void)n_in; (void)out_size; (void)ws_size;

  const float* q    = (const float*)d_in[0];
  const float* k_e  = (const float*)d_in[1];
  const float* k_i  = (const float*)d_in[2];
  const float* k_ei = (const float*)d_in[3];
  const float* v_e  = (const float*)d_in[4];
  const float* v_i  = (const float*)d_in[5];
  const float* v_ei = (const float*)d_in[6];
  const unsigned char* pmE  = (const unsigned char*)d_in[9];
  const unsigned char* pmEI = (const unsigned char*)d_in[10];
  const int* image_bool     = (const int*)d_in[11];
  const float* Wq   = (const float*)d_in[12]; const float* bq   = (const float*)d_in[13];
  const float* Wke  = (const float*)d_in[14]; const float* bke  = (const float*)d_in[15];
  const float* Wve  = (const float*)d_in[16]; const float* bve  = (const float*)d_in[17];
  const float* Wki  = (const float*)d_in[18]; const float* bki  = (const float*)d_in[19];
  const float* Wvi  = (const float*)d_in[20]; const float* bvi  = (const float*)d_in[21];
  const float* Wkei = (const float*)d_in[22]; const float* bkei = (const float*)d_in[23];
  const float* Wvei = (const float*)d_in[24]; const float* bvei = (const float*)d_in[25];
  const float* Wo   = (const float*)d_in[26]; const float* bo   = (const float*)d_in[27];
  float* out = (float*)d_out;

  const size_t NQ = (size_t)BATCH * SQ * DMODEL;   // 4M elems
  const size_t NI = (size_t)BATCH * SI * DMODEL;   // 1M elems
  bf16* ws  = (bf16*)d_ws;
  bf16* Qp  = ws;  ws += NQ;
  bf16* Ke  = ws;  ws += NQ;
  bf16* Ve  = ws;  ws += NQ;
  bf16* Ki  = ws;  ws += NI;
  bf16* Vi  = ws;  ws += NI;
  bf16* Kei = ws;  ws += NQ;
  bf16* Vei = ws;  ws += NQ;
  bf16* Cc  = ws;  ws += NQ;

  dim3 blk(256);
  dim3 g4(DMODEL / 128, (BATCH * SQ) / 64);   // (8, 64)
  dim3 g1(DMODEL / 128, (BATCH * SI) / 64);   // (8, 16)

  gemm_wmma<float, true><<<g4, blk, 0, stream>>>(q,    Wq,   bq,   Qp,  BATCH*SQ, DMODEL, DMODEL);
  gemm_wmma<float, true><<<g4, blk, 0, stream>>>(k_e,  Wke,  bke,  Ke,  BATCH*SQ, DMODEL, DMODEL);
  gemm_wmma<float, true><<<g4, blk, 0, stream>>>(v_e,  Wve,  bve,  Ve,  BATCH*SQ, DMODEL, DMODEL);
  gemm_wmma<float, true><<<g1, blk, 0, stream>>>(k_i,  Wki,  bki,  Ki,  BATCH*SI, DMODEL, DMODEL);
  gemm_wmma<float, true><<<g1, blk, 0, stream>>>(v_i,  Wvi,  bvi,  Vi,  BATCH*SI, DMODEL, DMODEL);
  gemm_wmma<float, true><<<g4, blk, 0, stream>>>(k_ei, Wkei, bkei, Kei, BATCH*SQ, DMODEL, DMODEL);
  gemm_wmma<float, true><<<g4, blk, 0, stream>>>(v_ei, Wvei, bvei, Vei, BATCH*SQ, DMODEL, DMODEL);

  // 4096 (b,h,qtile) waves, 8 per block
  attn_kernel<<<512, blk, 0, stream>>>(Qp, Ke, Ve, Ki, Vi, Kei, Vei,
                                       pmE, pmEI, image_bool, Cc);

  gemm_wmma<bf16, false><<<g4, blk, 0, stream>>>(Cc, Wo, bo, out, BATCH*SQ, DMODEL, DMODEL);
}
